// CrossModalAttentionFusion_82910048682651
// MI455X (gfx1250) — matrix operand order repaired and checked by reading, not compiled, not run on hardware
//
#include <hip/hip_runtime.h>
#include <hip/hip_bf16.h>

// ---------------------------------------------------------------------------
// CrossModalAttentionFusion, MI455X (gfx1250, wave32, WMMA + async LDS copy)
// softmax over a single key == 1  =>  head_out == V; Q/K are dead code.
// Pipeline: LN(x) -> GEMM(Wv^T)+bv -> GEMM(Wo^T)+bo -> LN -> out
// Weights are converted to bf16 AND transposed to [N][K] once, so both A and
// B tiles are K-contiguous and stream into LDS via global_load_async_to_lds.
// ---------------------------------------------------------------------------

typedef __attribute__((ext_vector_type(16))) __bf16 v16bf;
typedef __attribute__((ext_vector_type(8)))  float  v8f;
typedef int v4i __attribute__((ext_vector_type(4)));

#if defined(__has_builtin)
#if __has_builtin(__builtin_amdgcn_global_load_async_to_lds_b128)
#define USE_ASYNC_LDS 1
#endif
#endif
#ifndef USE_ASYNC_LDS
#define USE_ASYNC_LDS 0
#endif

#if USE_ASYNC_LDS
static __device__ __forceinline__ void async_cp16(const void* g, void* l) {
    // builtin signature: (v4i addrspace(1)*, v4i addrspace(3)*, Imm offset, Imm cpol)
    __builtin_amdgcn_global_load_async_to_lds_b128(
        (__attribute__((address_space(1))) v4i*)(g),
        (__attribute__((address_space(3))) v4i*)(l),
        0, 0);
}
static __device__ __forceinline__ void wait_async0() {
#if __has_builtin(__builtin_amdgcn_s_wait_asynccnt)
    __builtin_amdgcn_s_wait_asynccnt(0);
#else
    asm volatile("s_wait_asynccnt 0x0" ::: "memory");
#endif
}
#endif

static __device__ __forceinline__ unsigned short f2bf(float f) {
    unsigned int u = __float_as_uint(f);
    u += 0x7FFFu + ((u >> 16) & 1u);     // round-to-nearest-even
    return (unsigned short)(u >> 16);
}

// ------------- fp32 -> bf16 convert WITH [G][K][Ng] -> [G][Ng][K] transpose --
__global__ __launch_bounds__(256) void cvt_t_bf16(const float* __restrict__ in,
                                                  unsigned short* __restrict__ out,
                                                  int K, int Ng, int total) {
    int i = blockIdx.x * 256 + threadIdx.x;
    if (i >= total) return;
    int g = i / (K * Ng);
    int r = i - g * (K * Ng);
    int k = r / Ng;
    int n = r - k * Ng;
    out[(size_t)g * Ng * K + (size_t)n * K + k] = f2bf(in[i]);
}

// ----------------------------- row LayerNorm -------------------------------
template <bool OUT_BF16>
__global__ __launch_bounds__(256) void ln_kernel(const float* __restrict__ x,
                                                 const float* __restrict__ g,
                                                 const float* __restrict__ b,
                                                 void* __restrict__ out, int D) {
    const int row = blockIdx.x;
    const float* xr = x + (size_t)row * D;
    float s = 0.f, s2 = 0.f;
    for (int i = threadIdx.x; i < D; i += 256) {
        float v = xr[i];
        s += v; s2 += v * v;
    }
    for (int o = 16; o > 0; o >>= 1) {
        s  += __shfl_xor(s,  o);
        s2 += __shfl_xor(s2, o);
    }
    __shared__ float rs[8], rs2[8];
    const int w = threadIdx.x >> 5, lane = threadIdx.x & 31;
    if (lane == 0) { rs[w] = s; rs2[w] = s2; }
    __syncthreads();
    if (w == 0) {
        s  = (lane < 8) ? rs[lane]  : 0.f;
        s2 = (lane < 8) ? rs2[lane] : 0.f;
        for (int o = 4; o > 0; o >>= 1) {
            s  += __shfl_xor(s,  o);
            s2 += __shfl_xor(s2, o);
        }
        if (lane == 0) { rs[0] = s; rs2[0] = s2; }
    }
    __syncthreads();
    const float invD = 1.0f / (float)D;
    const float mean = rs[0] * invD;
    const float var  = rs2[0] * invD - mean * mean;
    const float rstd = rsqrtf(var + 1e-5f);
    for (int i = threadIdx.x; i < D; i += 256) {
        float y = (xr[i] - mean) * rstd * g[i] + b[i];
        if (OUT_BF16) ((unsigned short*)out)[(size_t)row * D + i] = f2bf(y);
        else          ((float*)out)[(size_t)row * D + i] = y;
    }
}

// ------------------------------ WMMA GEMM ----------------------------------
// C[M, Ntot] = A[M, K](bf16, row-major) * Bt[Ntot, K](bf16, K-major) + bias.
// Block tile 128x128, KC=64 (2 wmma k-steps), 256 thr = 8 waves as 4(M)x2(N),
// wave tile 32x64 -> 2x4 fragments, 16 WMMA per wave per LDS tile.
#define LDP 72   // padded LDS K-stride (bf16 elems): 64 + 8

template <bool OUT_BF16>
__global__ __launch_bounds__(256) void gemm_bf16_wmma(
    const unsigned short* __restrict__ A,      // [M, K]
    const unsigned short* __restrict__ Bt,     // [Ntot, K]
    const float* __restrict__ bias,            // [Ntot]
    void* __restrict__ Cout,                   // [M, Ntot]
    int M, int K, int Ntot) {

    __shared__ unsigned short sA[2][128 * LDP];  // [m][k]
    __shared__ unsigned short sB[2][128 * LDP];  // [n][k]

    const int m0 = blockIdx.x * 128;
    const int n0 = blockIdx.y * 128;

    const int tid  = threadIdx.x;
    const int lane = tid & 31;
    const int wv   = tid >> 5;
    const int wm   = wv & 3;          // 0..3  (M)
    const int wn   = wv >> 2;         // 0..1  (N)
    const int lh   = lane >> 4;       // lane half
    const int l16  = lane & 15;

    union Frag { uint4 q[2]; v16bf v; };

    v8f acc[2][4];
#pragma unroll
    for (int mi = 0; mi < 2; mi++)
#pragma unroll
        for (int ni = 0; ni < 4; ni++)
#pragma unroll
            for (int e = 0; e < 8; e++) acc[mi][ni][e] = 0.f;

#if USE_ASYNC_LDS
    // 128 rows x 64 k = 1024 chunks of 8 bf16 -> 4 chunks/thread per matrix
    auto loadtile = [&](int kt, int buf) {
        const int k0 = kt * 64;
#pragma unroll
        for (int i = 0; i < 4; i++) {
            const int c   = tid + 256 * i;
            const int row = c >> 3, kc = (c & 7) * 8;
            async_cp16(A  + (size_t)(m0 + row) * K + k0 + kc, &sA[buf][row * LDP + kc]);
            async_cp16(Bt + (size_t)(n0 + row) * K + k0 + kc, &sB[buf][row * LDP + kc]);
        }
    };
#else
    uint4 ra[4], rb[4];
    auto gload = [&](int kt) {
        const int k0 = kt * 64;
#pragma unroll
        for (int i = 0; i < 4; i++) {
            const int c   = tid + 256 * i;
            const int row = c >> 3, kc = (c & 7) * 8;
            ra[i] = *(const uint4*)(A  + (size_t)(m0 + row) * K + k0 + kc);
            rb[i] = *(const uint4*)(Bt + (size_t)(n0 + row) * K + k0 + kc);
        }
    };
    auto sstore = [&](int buf) {
#pragma unroll
        for (int i = 0; i < 4; i++) {
            const int c   = tid + 256 * i;
            const int row = c >> 3, kc = (c & 7) * 8;
            *(uint4*)&sA[buf][row * LDP + kc] = ra[i];
            *(uint4*)&sB[buf][row * LDP + kc] = rb[i];
        }
    };
#endif

    auto compute = [&](int buf) {
#pragma unroll
        for (int s = 0; s < 2; s++) {          // two 16x16x32 k-steps
            Frag fa[2], fb[4];
#pragma unroll
            for (int mi = 0; mi < 2; mi++) {
                const int row = wm * 32 + mi * 16 + l16;
                const uint4* pa = (const uint4*)&sA[buf][row * LDP];
                // ISA A layout: lane half h -> K = 8h..8h+7 and 16+8h..16+8h+7
                fa[mi].q[0] = pa[4 * s + lh];
                fa[mi].q[1] = pa[4 * s + 2 + lh];
            }
#pragma unroll
            for (int ni = 0; ni < 4; ni++) {
                const int nr = wn * 64 + ni * 16 + l16;
                const uint4* pb = (const uint4*)&sB[buf][nr * LDP];
                // B layout (matches A): lane half h -> K = 8h..8h+7 and 16+8h..16+8h+7 of column N=l16
                fb[ni].q[0] = pb[4 * s + lh];
                fb[ni].q[1] = pb[4 * s + 2 + lh];
            }
#pragma unroll
            for (int mi = 0; mi < 2; mi++)
#pragma unroll
                for (int ni = 0; ni < 4; ni++)
                    acc[mi][ni] = __builtin_amdgcn_wmma_f32_16x16x32_bf16(
                        false, fa[mi].v, false, fb[ni].v,
                        (short)0, acc[mi][ni], false, false);
        }
    };

    const int nk = K >> 6;  // K / 64

#if USE_ASYNC_LDS
    loadtile(0, 0);
    wait_async0();
    __syncthreads();
    for (int kt = 0;;) {
        if (kt + 1 < nk) loadtile(kt + 1, (kt + 1) & 1);
        compute(kt & 1);
        if (kt + 1 == nk) break;
        wait_async0();          // our async writes to the other buffer done
        __syncthreads();        // everyone's writes visible, reads finished
        ++kt;
    }
#else
    gload(0);
    sstore(0);
    __syncthreads();
    for (int kt = 0;;) {
        if (kt + 1 < nk) gload(kt + 1);
        compute(kt & 1);
        if (kt + 1 == nk) break;
        sstore((kt + 1) & 1);
        __syncthreads();
        ++kt;
    }
#endif

    // epilogue: C[r + 8*lh][l16] per fragment (ISA C/D layout), + bias
#pragma unroll
    for (int mi = 0; mi < 2; mi++)
#pragma unroll
        for (int ni = 0; ni < 4; ni++) {
            const int gn = n0 + wn * 64 + ni * 16 + l16;
            const float bn = bias[gn];
#pragma unroll
            for (int r = 0; r < 8; r++) {
                const int gm = m0 + wm * 32 + mi * 16 + 8 * lh + r;
                const float val = acc[mi][ni][r] + bn;
                if (OUT_BF16)
                    ((unsigned short*)Cout)[(size_t)gm * Ntot + gn] = f2bf(val);
                else
                    ((float*)Cout)[(size_t)gm * Ntot + gn] = val;
            }
        }
}

// ------------------------------- launcher ----------------------------------
extern "C" void kernel_launch(void* const* d_in, const int* in_sizes, int n_in,
                              void* d_out, int out_size, void* d_ws, size_t ws_size,
                              hipStream_t stream) {
    (void)in_sizes; (void)n_in; (void)out_size; (void)ws_size;

    constexpr int B   = 16384;
    constexpr int XC  = 2048;
    constexpr int OUT = 1024;
    constexpr int WV_N = 8 * 2048 * 128;   // 2,097,152
    constexpr int WO_N = 1024 * 1024;      // 1,048,576

    const float* x_in = (const float*)d_in[1];   // xception_features
    const float* Wv   = (const float*)d_in[6];
    const float* bv   = (const float*)d_in[7];
    const float* Wo   = (const float*)d_in[8];
    const float* bo   = (const float*)d_in[9];
    const float* g2   = (const float*)d_in[12];
    const float* b2   = (const float*)d_in[13];
    const float* g3   = (const float*)d_in[14];
    const float* b3   = (const float*)d_in[15];
    float* out = (float*)d_out;

    char* ws = (char*)d_ws;
    unsigned short* wv_t  = (unsigned short*)(ws);                     //  4,194,304 B  [1024][2048]
    unsigned short* wo_t  = (unsigned short*)(ws + 4194304);           //  2,097,152 B  [1024][1024]
    unsigned short* xln   = (unsigned short*)(ws + 6291456);           // 67,108,864 B
    unsigned short* vbuf  = (unsigned short*)(ws + 73400320);          // 33,554,432 B
    float*          fused = (float*)(ws + 6291456);                    // reuse xln region

    // 1) weight downconvert + transpose to [N][K]
    //    Wv [8][2048][128] -> [8][128][2048] == [1024][2048]
    cvt_t_bf16<<<(WV_N + 255) / 256, 256, 0, stream>>>(Wv, wv_t, XC, 128, WV_N);
    //    Wo [1024][1024]   -> [1024][1024] transposed
    cvt_t_bf16<<<(WO_N + 255) / 256, 256, 0, stream>>>(Wo, wo_t, OUT, OUT, WO_N);

    // 2) LN(xception) -> bf16
    ln_kernel<true><<<B, 256, 0, stream>>>(x_in, g2, b2, xln, XC);

    // 3) V = xln @ Wv + bv
    gemm_bf16_wmma<true><<<dim3(B / 128, OUT / 128), 256, 0, stream>>>(
        xln, wv_t, bv, vbuf, B, XC, OUT);

    // 4) fused = V @ Wo + bo
    gemm_bf16_wmma<false><<<dim3(B / 128, OUT / 128), 256, 0, stream>>>(
        vbuf, wo_t, bo, fused, B, OUT, OUT);

    // 5) LN(fused) -> out (fp32)
    ln_kernel<false><<<B, 256, 0, stream>>>(fused, g3, b3, out, OUT);
}
